// BiLSTMCRF_29987461661098
// MI455X (gfx1250) — matrix operand order repaired and checked by reading, not compiled, not run on hardware
//
#include <hip/hip_runtime.h>
#include <hip/hip_bf16.h>
#include <stdint.h>

// ---------------- problem constants ----------------
constexpr int Bb = 128;       // batch
constexpr int Tt = 512;       // sequence length
constexpr int Dd = 500;       // embedding dim
constexpr int Dp = 512;       // padded embedding dim (K multiple of 32)
constexpr int Hh = 256;       // hidden per direction
constexpr int Gg = 1024;      // 4*H gate rows
constexpr int Ll = 25;        // labels
constexpr int Lp = 32;        // padded labels
constexpr int BT = Bb * Tt;   // 65536 positions
constexpr int NSLICE = 16;    // persistent workgroups in the recurrent kernel
constexpr int WST = 264;      // padded LDS row stride (halves) for Whh slices:
                              // 264*2B = 528B = 132 dwords, 132 % 64 = 4 -> banks spread

typedef __attribute__((ext_vector_type(8)))  __bf16 v8bf;
typedef __attribute__((ext_vector_type(16))) __bf16 v16bf;
typedef __attribute__((ext_vector_type(8)))  float  v8f;

static __device__ __forceinline__ v16bf cat16(v8bf lo, v8bf hi) {
  return __builtin_shufflevector(lo, hi, 0,1,2,3,4,5,6,7,8,9,10,11,12,13,14,15);
}

// A fragment: 16x32 bf16 (MxK) from row-major [m][k].
// Lane L: row = L&15, K chunks [hi*8,+8) and [16+hi*8,+8)  (ISA 7.12.2).
static __device__ __forceinline__ v16bf load_a_frag(const __bf16* base, int lda, int kb) {
  int lane = threadIdx.x & 31;
  int m = lane & 15, hi = lane >> 4;
  const __bf16* p = base + (size_t)m * lda + kb + hi * 8;
  return cat16(*(const v8bf*)p, *(const v8bf*)(p + 16));
}

// B fragment: 32x16 bf16 (KxN) built from row-major weights [n][k] (B = W^T).
// Lane L: col n = L&15, K chunks [hi*8,+8) and [16+hi*8,+8) (matches A).
// Works for global or LDS-resident weights (addrspace inferred -> ds_load_b128).
static __device__ __forceinline__ v16bf load_b_frag(const __bf16* base, int ldb, int kb) {
  int lane = threadIdx.x & 31;
  int n = lane & 15, hi = lane >> 4;
  const __bf16* p = base + (size_t)n * ldb + kb + hi * 8;
  return cat16(*(const v8bf*)p, *(const v8bf*)(p + 16));
}

static __device__ __forceinline__ v8f wmma_bf16(v16bf a, v16bf b, v8f c) {
  // (neg_a, A, neg_b, B, c_mod, C, reuse_a, reuse_b)
  return __builtin_amdgcn_wmma_f32_16x16x32_bf16(false, a, false, b, (short)0, c, false, false);
}

static __device__ __forceinline__ float sigm(float x) { return 1.f / (1.f + __expf(-x)); }

// ---------------- prep: fp32 -> bf16 with zero padding ----------------
__global__ void k_cvt_pad(const float* __restrict__ src, __bf16* __restrict__ dst,
                          int srcRows, int srcCols, int dstCols, long long total) {
  for (long long i = (long long)blockIdx.x * blockDim.x + threadIdx.x; i < total;
       i += (long long)gridDim.x * blockDim.x) {
    int r = (int)(i / dstCols), c = (int)(i % dstCols);
    float v = (r < srcRows && c < srcCols) ? src[(size_t)r * srcCols + c] : 0.f;
    dst[i] = (__bf16)v;
  }
}

__global__ void k_lengths(const float* __restrict__ mask, int* __restrict__ lengths) {
  int b = threadIdx.x;
  if (b < Bb) {
    float s = 0.f;
    for (int t = 0; t < Tt; ++t) s += mask[b * Tt + t];
    lengths[b] = (int)(s + 0.5f);
  }
}

// ---------------- K1: xw = X(bf16 MxDp) * W(bf16 GgxDp)^T + bias, bf16 out ----------------
// block = 256 (8 waves); wave = one 16-row M tile over a 64-wide N tile.
// Software-pipelined k-loop: next fragments prefetched into registers before WMMAs.
__global__ void __launch_bounds__(256) k_gemm_xw(const __bf16* __restrict__ X,
                                                 const __bf16* __restrict__ W,
                                                 const float* __restrict__ bias,
                                                 __bf16* __restrict__ out) {
  int wv = threadIdx.x >> 5, lane = threadIdx.x & 31;
  int mbase = blockIdx.x * 128 + wv * 16;
  int nbase = blockIdx.y * 64;
  v8f z8 = {0.f,0.f,0.f,0.f,0.f,0.f,0.f,0.f};
  v8f acc[4] = {z8, z8, z8, z8};
  const __bf16* Arow = X + (size_t)mbase * Dp;
  v16bf a = load_a_frag(Arow, Dp, 0);
  v16bf bf[4];
#pragma unroll
  for (int j = 0; j < 4; ++j) bf[j] = load_b_frag(W + (size_t)(nbase + j * 16) * Dp, Dp, 0);
#pragma unroll
  for (int kb = 0; kb < Dp; kb += 32) {
    v16bf a_n = a;
    v16bf bf_n[4] = {bf[0], bf[1], bf[2], bf[3]};
    if (kb + 32 < Dp) {
      a_n = load_a_frag(Arow, Dp, kb + 32);
#pragma unroll
      for (int j = 0; j < 4; ++j)
        bf_n[j] = load_b_frag(W + (size_t)(nbase + j * 16) * Dp, Dp, kb + 32);
    }
#pragma unroll
    for (int j = 0; j < 4; ++j) acc[j] = wmma_bf16(a, bf[j], acc[j]);
    a = a_n;
#pragma unroll
    for (int j = 0; j < 4; ++j) bf[j] = bf_n[j];
  }
  int nl = lane & 15, hi = lane >> 4;
#pragma unroll
  for (int j = 0; j < 4; ++j)
#pragma unroll
    for (int r = 0; r < 8; ++r) {
      int m = mbase + r + hi * 8;
      int n = nbase + j * 16 + nl;
      out[(size_t)m * Gg + n] = (__bf16)(acc[j][r] + bias[n]);
    }
}

// ---------------- K2: persistent bidirectional LSTM recurrence ----------------
// 16 WGs x 512 threads. WG `slice` owns hidden units [slice*16, slice*16+16) for BOTH
// directions. Per step:
//   (a) async-gather this step's xw gate slice into LDS (overlaps with the GEMM),
//   (b) WMMA GEMM h_prev x Whh_slice^T with Whh staged in LDS (loop-invariant),
//   (c) wait asynccnt + barrier, elementwise cell update, write h (bf16) for next step.
// Steps separated by an agent-scope atomic grid barrier; h double-buffered in global L2.
__global__ void __launch_bounds__(512) k_lstm(
    const __bf16* __restrict__ xw_f, const __bf16* __restrict__ xw_b,
    const __bf16* __restrict__ Whh_f, const __bf16* __restrict__ Whh_b,
    __bf16* __restrict__ h_buf,     // [parity][dir][B][H]
    __bf16* __restrict__ hseq,      // [B][T][2H]
    const int* __restrict__ lengths,
    int* __restrict__ cnt) {
  __shared__ float zslab[128 * 64];                        // [b][gate][16]   32KB
  __shared__ float cst[2][128 * 16];                       // c state per dir 16KB
  __shared__ __align__(16) __bf16 xwslab[128 * 64];        // staged xw slice 16KB
  __shared__ __align__(16) __bf16 wlds[2][64 * WST];       // Whh slices      ~66KB
  const int slice = blockIdx.x;
  const int tid = threadIdx.x;
  const int wv = tid >> 5, lane = tid & 31;
  const int mt = wv >> 1, gp = wv & 1;          // 8 M-tiles x 2 gate-pairs = 16 waves
  const int nl = lane & 15, hi = lane >> 4;

  for (int i = tid; i < 2 * 128 * 16; i += 512) ((float*)cst)[i] = 0.f;
  // stage loop-invariant Whh slices (rows g*256+slice*16+r, all K) into padded LDS
  for (int i = tid; i < 2 * 64 * 32; i += 512) {
    int d = i >> 11;            // direction
    int rrow = (i >> 5) & 63;   // local row 0..63 (= g*16 + r)
    int c = i & 31;             // 8-half chunk
    int g = rrow >> 4, r = rrow & 15;
    const __bf16* src = (d ? Whh_b : Whh_f) + (size_t)(g * Hh + slice * 16 + r) * Hh + c * 8;
    *(v8bf*)&wlds[d][rrow * WST + c * 8] = *(const v8bf*)src;
  }
  __syncthreads();

  const uint32_t xws_base = (uint32_t)(uintptr_t)&xwslab[0];

  for (int t = 0; t < Tt; ++t) {
    for (int dir = 0; dir < 2; ++dir) {
      const __bf16* xw = dir ? xw_b : xw_f;
      const __bf16* hprev = h_buf + (size_t)((t & 1) * 2 + dir) * (Bb * Hh);
      __bf16* hnext = h_buf + (size_t)(((t + 1) & 1) * 2 + dir) * (Bb * Hh);

      // (a) async gather of this step's xw gate slice into LDS (ASYNCcnt path);
      //     overlaps with the recurrent GEMM below.
#pragma unroll
      for (int r2 = 0; r2 < 2; ++r2) {
        int item = tid + r2 * 512;          // 1024 b128 transfers: (b, gate, half)
        int b = item >> 3;
        int g = (item >> 1) & 3;
        int half = item & 1;
        int len = lengths[b];
        int pos = dir ? ((t < len) ? (len - 1 - t) : t) : t;   // rev_idx gather
        const __bf16* gsrc = xw + ((size_t)b * Tt + pos) * Gg + g * Hh + slice * 16 + half * 8;
        uint32_t la = xws_base + (uint32_t)((b * 64 + g * 16 + half * 8) * 2);
        asm volatile("global_load_async_to_lds_b128 %0, %1, off"
                     :: "v"(la), "v"((uint64_t)(uintptr_t)gsrc)
                     : "memory");
      }

      // (b) Zslice[128 x 64] = h_prev[128x256] * Whh_rows^T (4 gate blocks of 16),
      //     A from global (pipelined), B from LDS.
      v8f z8 = {0.f,0.f,0.f,0.f,0.f,0.f,0.f,0.f};
      v8f acc0 = z8, acc1 = z8;
      const __bf16* Arow = hprev + (size_t)(mt * 16) * Hh;
      const __bf16* B0 = &wlds[dir][(gp * 2 + 0) * 16 * WST];
      const __bf16* B1 = &wlds[dir][(gp * 2 + 1) * 16 * WST];
      v16bf a = load_a_frag(Arow, Hh, 0);
#pragma unroll
      for (int kb = 0; kb < Hh; kb += 32) {
        v16bf a_n = a;
        if (kb + 32 < Hh) a_n = load_a_frag(Arow, Hh, kb + 32);
        acc0 = wmma_bf16(a, load_b_frag(B0, WST, kb), acc0);
        acc1 = wmma_bf16(a, load_b_frag(B1, WST, kb), acc1);
        a = a_n;
      }
#pragma unroll
      for (int r = 0; r < 8; ++r) {
        int m = mt * 16 + r + hi * 8;
        zslab[m * 64 + (gp * 2 + 0) * 16 + nl] = acc0[r];
        zslab[m * 64 + (gp * 2 + 1) * 16 + nl] = acc1[r];
      }
      // (c) all async xw transfers landed + zslab visible to whole WG
      asm volatile("s_wait_asynccnt 0x0" ::: "memory");
      __syncthreads();

      // elementwise LSTM cell update for this WG's 128x16 (batch x hidden) slice
#pragma unroll
      for (int r = 0; r < 4; ++r) {
        int idx = tid + r * 512;
        int b = idx >> 4, n = idx & 15;
        int len = lengths[b];
        int pos = dir ? ((t < len) ? (len - 1 - t) : t) : t;
        size_t row = (size_t)b * Tt + pos;
        float zi = zslab[b * 64 +  0 + n] + (float)xwslab[b * 64 +  0 + n];
        float zf = zslab[b * 64 + 16 + n] + (float)xwslab[b * 64 + 16 + n];
        float zg = zslab[b * 64 + 32 + n] + (float)xwslab[b * 64 + 32 + n];
        float zo = zslab[b * 64 + 48 + n] + (float)xwslab[b * 64 + 48 + n];
        float c = sigm(zf) * cst[dir][b * 16 + n] + sigm(zi) * tanhf(zg);
        float h = sigm(zo) * tanhf(c);
        cst[dir][b * 16 + n] = c;
        hnext[(size_t)b * Hh + slice * 16 + n] = (__bf16)h;
        hseq[row * (2 * Hh) + dir * Hh + slice * 16 + n] = (__bf16)h;
        if (t + 1 < Tt) {   // warm L2/GL1 for next step's async gather
          int npos = dir ? ((t + 1 < len) ? (len - 2 - t) : (t + 1)) : (t + 1);
          __builtin_prefetch(xw + ((size_t)b * Tt + npos) * Gg + slice * 16 + n, 0, 0);
        }
      }
      __syncthreads();
    }
    // grid-wide barrier across the 16 persistent WGs (monotonic counter)
    if (tid == 0) {
      __hip_atomic_fetch_add(cnt, 1, __ATOMIC_ACQ_REL, __HIP_MEMORY_SCOPE_AGENT);
      const int target = NSLICE * (t + 1);
      while (__hip_atomic_load(cnt, __ATOMIC_ACQUIRE, __HIP_MEMORY_SCOPE_AGENT) < target)
        __builtin_amdgcn_s_sleep(2);
    }
    __syncthreads();
  }
}

// ---------------- K3: emissions em = (hseq * linW^T + linb) * mask ----------------
__global__ void __launch_bounds__(256) k_gemm_em(
    const __bf16* __restrict__ hseq, const __bf16* __restrict__ Wl,
    const float* __restrict__ linb, const float* __restrict__ mask,
    float* __restrict__ em) {
  int wv = threadIdx.x >> 5, lane = threadIdx.x & 31;
  int mbase = blockIdx.x * 128 + wv * 16;
  v8f z8 = {0.f,0.f,0.f,0.f,0.f,0.f,0.f,0.f};
  v8f acc0 = z8, acc1 = z8;
  const __bf16* Arow = hseq + (size_t)mbase * (2 * Hh);
  v16bf a  = load_a_frag(Arow, 2 * Hh, 0);
  v16bf b0 = load_b_frag(Wl, 2 * Hh, 0);
  v16bf b1 = load_b_frag(Wl + (size_t)16 * (2 * Hh), 2 * Hh, 0);
#pragma unroll
  for (int kb = 0; kb < 2 * Hh; kb += 32) {
    v16bf a_n = a, b0_n = b0, b1_n = b1;
    if (kb + 32 < 2 * Hh) {
      a_n  = load_a_frag(Arow, 2 * Hh, kb + 32);
      b0_n = load_b_frag(Wl, 2 * Hh, kb + 32);
      b1_n = load_b_frag(Wl + (size_t)16 * (2 * Hh), 2 * Hh, kb + 32);
    }
    acc0 = wmma_bf16(a, b0, acc0);
    acc1 = wmma_bf16(a, b1, acc1);
    a = a_n; b0 = b0_n; b1 = b1_n;
  }
  int nl = lane & 15, hi = lane >> 4;
#pragma unroll
  for (int r = 0; r < 8; ++r) {
    int row = mbase + r + hi * 8;
    float mk = mask[row];
    int n0 = nl, n1 = 16 + nl;
    if (n0 < Ll) em[(size_t)row * Ll + n0] = (acc0[r] + linb[n0]) * mk;
    if (n1 < Ll) em[(size_t)row * Ll + n1] = (acc1[r] + linb[n1]) * mk;
  }
}

// ---------------- K4: CRF log-likelihood (forward algorithm + numerator) ----------------
// One wave per batch row; lane j owns alpha[j]; trans cached in LDS.
__global__ void __launch_bounds__(256) k_crf(
    const float* __restrict__ em, const float* __restrict__ mask,
    const int* __restrict__ labels, const int* __restrict__ lengths,
    const float* __restrict__ trans, const float* __restrict__ strans,
    const float* __restrict__ etrans, float* __restrict__ ll) {
  __shared__ float tr[Ll * Ll];
  __shared__ float ab[8][Lp];
  for (int i = threadIdx.x; i < Ll * Ll; i += 256) tr[i] = trans[i];
  int wv = threadIdx.x >> 5, j = threadIdx.x & 31;
  int jc = (j < Ll) ? j : (Ll - 1);
  int b = blockIdx.x * 8 + wv;
  float alpha = (j < Ll) ? strans[j] + em[(size_t)b * Tt * Ll + j] : -1e30f;
  ab[wv][j] = alpha;
  __syncthreads();
  for (int t = 1; t < Tt; ++t) {
    float emt = em[((size_t)b * Tt + t) * Ll + jc];
    float mk = mask[b * Tt + t];
    float mx = -1e30f;
    for (int i = 0; i < Ll; ++i) mx = fmaxf(mx, ab[wv][i] + tr[i * Ll + jc]);
    float s = 0.f;
    for (int i = 0; i < Ll; ++i) s += __expf(ab[wv][i] + tr[i * Ll + jc] - mx);
    float nj = mx + __logf(s) + emt;
    if (mk > 0.f && j < Ll) alpha = nj;
    __syncthreads();
    ab[wv][j] = alpha;
    __syncthreads();
  }
  float mx = -1e30f;
  for (int i = 0; i < Ll; ++i) mx = fmaxf(mx, ab[wv][i] + etrans[i]);
  float s = 0.f;
  for (int i = 0; i < Ll; ++i) s += __expf(ab[wv][i] + etrans[i] - mx);
  float logZ = mx + __logf(s);
  if (j == 0) {
    const int* lab = labels + b * Tt;
    const float* mkp = mask + b * Tt;
    int len = lengths[b];
    float num = strans[lab[0]] + em[(size_t)b * Tt * Ll + lab[0]] * mkp[0];
    int prev = lab[0];
    for (int t = 1; t < Tt; ++t) {
      int l = lab[t];
      float mk = mkp[t];
      num += em[((size_t)b * Tt + t) * Ll + l] * mk + tr[prev * Ll + l] * mk;
      prev = l;
    }
    num += etrans[lab[len - 1]];
    ll[b] = num - logZ;
  }
}

// ---------------- K5a: Viterbi scan (delta + backpointers) ----------------
__global__ void __launch_bounds__(256) k_viterbi(
    const float* __restrict__ em, const float* __restrict__ mask,
    const float* __restrict__ trans, const float* __restrict__ strans,
    unsigned char* __restrict__ bp, float* __restrict__ dfin) {
  __shared__ float tr[Ll * Ll];
  __shared__ float db[8][Lp];
  for (int i = threadIdx.x; i < Ll * Ll; i += 256) tr[i] = trans[i];
  int wv = threadIdx.x >> 5, j = threadIdx.x & 31;
  int jc = (j < Ll) ? j : (Ll - 1);
  int b = blockIdx.x * 8 + wv;
  float delta = (j < Ll) ? strans[j] + em[(size_t)b * Tt * Ll + j] : -1e30f;
  db[wv][j] = delta;
  __syncthreads();
  for (int t = 1; t < Tt; ++t) {
    float emt = em[((size_t)b * Tt + t) * Ll + jc];
    float mk = mask[b * Tt + t];
    float best = -1e30f; int arg = 0;
    for (int i = 0; i < Ll; ++i) {
      float v = db[wv][i] + tr[i * Ll + jc];
      if (v > best) { best = v; arg = i; }   // first-max tie-break like jnp.argmax
    }
    bool keep = mk > 0.f;
    if (j < Ll) {
      if (keep) delta = best + emt;
      bp[((size_t)(t - 1) * Bb + b) * Ll + j] = (unsigned char)(keep ? arg : j);
    }
    __syncthreads();
    db[wv][j] = delta;
    __syncthreads();
  }
  if (j < Ll) dfin[b * Ll + j] = delta;
}

// ---------------- K5b: backtrace + masked path write ----------------
__global__ void k_backtrace(const float* __restrict__ dfin, const float* __restrict__ etrans,
                            const unsigned char* __restrict__ bp, const float* __restrict__ mask,
                            float* __restrict__ preds) {
  int b = threadIdx.x;
  if (b >= Bb) return;
  float best = -1e30f; int cur = 0;
  for (int jj = 0; jj < Ll; ++jj) {
    float v = dfin[b * Ll + jj] + etrans[jj];
    if (v > best) { best = v; cur = jj; }
  }
  preds[(size_t)b * Tt + (Tt - 1)] = (mask[b * Tt + Tt - 1] > 0.f) ? (float)cur : 0.f;
  for (int t = Tt - 1; t >= 1; --t) {
    cur = bp[((size_t)(t - 1) * Bb + b) * Ll + cur];
    preds[(size_t)b * Tt + (t - 1)] = (mask[b * Tt + t - 1] > 0.f) ? (float)cur : 0.f;
  }
}

// ---------------- K6: loss = -mean(ll), deterministic overwrite ----------------
__global__ void k_loss(const float* __restrict__ ll, float* __restrict__ out) {
  __shared__ float red[128];
  int i = threadIdx.x;
  red[i] = ll[i];
  __syncthreads();
  for (int s = 64; s > 0; s >>= 1) {
    if (i < s) red[i] += red[i + s];
    __syncthreads();
  }
  if (i == 0) out[0] = -red[0] / (float)Bb;
}

// ---------------- host orchestration ----------------
extern "C" void kernel_launch(void* const* d_in, const int* in_sizes, int n_in,
                              void* d_out, int out_size, void* d_ws, size_t ws_size,
                              hipStream_t stream) {
  const float* input_embed = (const float*)d_in[0];
  const float* mask        = (const float*)d_in[1];
  const int*   labels      = (const int*)d_in[2];
  const float* Wih_f = (const float*)d_in[3];
  const float* Whh_f = (const float*)d_in[4];
  const float* b_f   = (const float*)d_in[5];
  const float* Wih_b = (const float*)d_in[6];
  const float* Whh_b = (const float*)d_in[7];
  const float* b_b   = (const float*)d_in[8];
  const float* linW  = (const float*)d_in[9];
  const float* linb  = (const float*)d_in[10];
  const float* trans = (const float*)d_in[11];
  const float* strans = (const float*)d_in[12];
  const float* etrans = (const float*)d_in[13];

  char* ws = (char*)d_ws;
  size_t off = 0;
  auto alloc = [&](size_t bytes) -> size_t {
    size_t p = off;
    off = (off + bytes + 255) & ~(size_t)255;
    return p;
  };
  int*    lengths = (int*)   (ws + alloc((size_t)Bb * 4));
  float*  ll      = (float*) (ws + alloc((size_t)Bb * 4));
  int*    cnt     = (int*)   (ws + alloc(256));
  float*  dfin    = (float*) (ws + alloc((size_t)Bb * Ll * 4));
  __bf16* Xbf     = (__bf16*)(ws + alloc((size_t)BT * Dp * 2));      // 64 MB
  __bf16* WihFbf  = (__bf16*)(ws + alloc((size_t)Gg * Dp * 2));
  __bf16* WihBbf  = (__bf16*)(ws + alloc((size_t)Gg * Dp * 2));
  __bf16* WhhFbf  = (__bf16*)(ws + alloc((size_t)Gg * Hh * 2));
  __bf16* WhhBbf  = (__bf16*)(ws + alloc((size_t)Gg * Hh * 2));
  __bf16* WlBf    = (__bf16*)(ws + alloc((size_t)Lp * 2 * Hh * 2));
  __bf16* xwF     = (__bf16*)(ws + alloc((size_t)BT * Gg * 2));      // 128 MB
  __bf16* xwB     = (__bf16*)(ws + alloc((size_t)BT * Gg * 2));      // 128 MB
  __bf16* hseq    = (__bf16*)(ws + alloc((size_t)BT * 2 * Hh * 2));  // 64 MB
  __bf16* hbuf    = (__bf16*)(ws + alloc((size_t)2 * 2 * Bb * Hh * 2));
  float*  em      = (float*) (ws + alloc((size_t)BT * Ll * 4));
  unsigned char* bp = (unsigned char*)(ws + alloc((size_t)(Tt - 1) * Bb * Ll));
  // total workspace ~415 MB

  hipMemsetAsync(cnt, 0, 4, stream);
  hipMemsetAsync(hbuf, 0, (size_t)2 * 2 * Bb * Hh * 2, stream);

  // precision prep (fp32 -> bf16, zero-padded K)
  k_cvt_pad<<<4096, 256, 0, stream>>>(input_embed, Xbf, BT, Dd, Dp, (long long)BT * Dp);
  k_cvt_pad<<<512, 256, 0, stream>>>(Wih_f, WihFbf, Gg, Dd, Dp, (long long)Gg * Dp);
  k_cvt_pad<<<512, 256, 0, stream>>>(Wih_b, WihBbf, Gg, Dd, Dp, (long long)Gg * Dp);
  k_cvt_pad<<<256, 256, 0, stream>>>(Whh_f, WhhFbf, Gg, Hh, Hh, (long long)Gg * Hh);
  k_cvt_pad<<<256, 256, 0, stream>>>(Whh_b, WhhBbf, Gg, Hh, Hh, (long long)Gg * Hh);
  k_cvt_pad<<<64, 256, 0, stream>>>(linW, WlBf, Ll, 2 * Hh, 2 * Hh, (long long)Lp * 2 * Hh);
  k_lengths<<<1, 128, 0, stream>>>(mask, lengths);

  // big parallel input GEMMs (bias baked in)
  dim3 gx(BT / 128, Gg / 64);
  k_gemm_xw<<<gx, 256, 0, stream>>>(Xbf, WihFbf, b_f, xwF);
  k_gemm_xw<<<gx, 256, 0, stream>>>(Xbf, WihBbf, b_b, xwB);

  // persistent bidirectional recurrence (16 WGs, software grid barrier per step)
  k_lstm<<<NSLICE, 512, 0, stream>>>(xwF, xwB, WhhFbf, WhhBbf, hbuf, hseq, lengths, cnt);

  // emissions + CRF + Viterbi
  k_gemm_em<<<BT / 128, 256, 0, stream>>>(hseq, WlBf, linb, mask, em);
  k_crf<<<Bb / 8, 256, 0, stream>>>(em, mask, labels, lengths, trans, strans, etrans, ll);
  k_viterbi<<<Bb / 8, 256, 0, stream>>>(em, mask, trans, strans, bp, dfin);

  float* out = (float*)d_out;
  k_backtrace<<<1, 128, 0, stream>>>(dfin, etrans, bp, mask, out + 1);
  k_loss<<<1, 128, 0, stream>>>(ll, out);
}
